// GateRNN_3710851744208
// MI455X (gfx1250) — compile-verified, matrix operand order checked by reading, not run on hardware
//
#include <hip/hip_runtime.h>
#include <hip/hip_bf16.h>

#define B_ 32
#define T_ 256
#define IN_ 64
#define D_ 128

typedef __attribute__((ext_vector_type(16))) __bf16 v16bf;
typedef __attribute__((ext_vector_type(8)))  __bf16 v8bf;
typedef __attribute__((ext_vector_type(8)))  float  v8f;

// ---------- helpers ----------
static __device__ __forceinline__ __bf16 f2bf(float f) {
    union { float f; unsigned u; } v; v.f = f;
    unsigned r = (v.u + 0x7FFFu + ((v.u >> 16) & 1u)) >> 16;
    unsigned short s = (unsigned short)r;
    __bf16 o; __builtin_memcpy(&o, &s, 2); return o;
}
static __device__ __forceinline__ float gelu_exact(float x) {
    return 0.5f * x * (1.0f + erff(x * 0.70710678118654752f));
}
static __device__ __forceinline__ float dgelu_exact(float x) {
    float Phi = 0.5f * (1.0f + erff(x * 0.70710678118654752f));
    float phi = 0.39894228040143268f * expf(-0.5f * x * x);
    return Phi + x * phi;
}
static __device__ __forceinline__ float sigm(float x) { return 1.0f / (1.0f + expf(-x)); }

// ---------- 1) Wgate -> transposed bf16 (WMMA B operand source) ----------
__global__ void prep_wgt_kernel(const float* __restrict__ Wgate, __bf16* __restrict__ WgT) {
    int idx = blockIdx.x * 256 + threadIdx.x;          // 64 blocks x 256 = 16384
    int i = idx >> 7, k = idx & 127;
    WgT[idx] = f2bf(Wgate[k * D_ + i]);                // WgT[i][k] = Wgate[k][i]
}

// ---------- 2) forward scan: one block per batch, thread d owns state lane ----------
__global__ void fwd_kernel(const float* __restrict__ x, const float* __restrict__ h0,
                           const float* __restrict__ Wx, const float* __restrict__ Wgate,
                           const float* __restrict__ Wh,
                           float* __restrict__ hfin, float* __restrict__ y,
                           float* __restrict__ gbuf, float* __restrict__ dphibuf) {
    const int b = blockIdx.x;
    const int d = threadIdx.x;                 // 0..127
    __shared__ float hs[D_];
    __shared__ float as_[D_];
    hs[d] = h0[b * D_ + d];
    __syncthreads();

    for (int t = 0; t < T_; ++t) {
        const float* xr = x + ((size_t)b * T_ + t) * IN_;
        float pre = 0.0f;
        #pragma unroll 8
        for (int j = 0; j < IN_; ++j) pre = fmaf(xr[j], Wx[j * D_ + d], pre);
        #pragma unroll 8
        for (int j = 0; j < D_; ++j)  pre = fmaf(hs[j], Wh[j * D_ + d], pre);
        float a   = gelu_exact(pre);
        float dph = dgelu_exact(pre);
        as_[d] = a;
        __syncthreads();
        float s = 0.0f;
        #pragma unroll 8
        for (int k = 0; k < D_; ++k) s = fmaf(as_[k], Wgate[k * D_ + d], s);
        float g  = sigm(s);
        float hn = g * hs[d] + (1.0f - g);

        size_t tb = (size_t)t * B_ + b;
        gbuf[tb * D_ + d]    = g;
        dphibuf[tb * D_ + d] = dph;
        y[((size_t)b * T_ + t) * D_ + d] = hn;
        __syncthreads();
        hs[d] = hn;
        __syncthreads();
    }
    hfin[b * D_ + d] = hs[d];
}

// ---------- 3) final-step factors for dh_dWh ----------
__global__ void finstep_kernel(const float* __restrict__ x, const float* __restrict__ hfin,
                               const float* __restrict__ Wx, const float* __restrict__ Wgate,
                               const float* __restrict__ Wh,
                               float* __restrict__ coefF, float* __restrict__ dphiF) {
    const int b = blockIdx.x;
    const int d = threadIdx.x;
    __shared__ float hs[D_];
    __shared__ float as_[D_];
    hs[d] = hfin[b * D_ + d];
    __syncthreads();
    const float* xr = x + ((size_t)b * T_ + (T_ - 1)) * IN_;
    float pre = 0.0f;
    for (int j = 0; j < IN_; ++j) pre = fmaf(xr[j], Wx[j * D_ + d], pre);
    for (int j = 0; j < D_; ++j)  pre = fmaf(hs[j], Wh[j * D_ + d], pre);
    float a = gelu_exact(pre);
    as_[d] = a;
    __syncthreads();
    float s = 0.0f;
    for (int k = 0; k < D_; ++k) s = fmaf(as_[k], Wgate[k * D_ + d], s);
    float g = sigm(s);
    coefF[b * D_ + d] = (hs[d] - 1.0f) * g * (1.0f - g);
    dphiF[b * D_ + d] = dgelu_exact(pre);
}

// ---------- 4) Jacobians: one block per (t,b); 8 waves; bf16 WMMA 128x128x128 ----------
__global__ void __launch_bounds__(256, 2)
jac_kernel(const float* __restrict__ Wh, const __bf16* __restrict__ WgT,
           const float* __restrict__ gbuf, const float* __restrict__ dphibuf,
           const float* __restrict__ y, const float* __restrict__ h0,
           float* __restrict__ jac_out) {
    const int tb  = blockIdx.x;          // t*B + b
    const int t   = tb >> 5;
    const int b   = tb & 31;
    const int tid = threadIdx.x;         // 0..255

    __shared__ __bf16 As[D_ * D_];       // A[j][k] = bf16(Wh[j][k] * dphi[k])
    __shared__ __bf16 Bs[D_ * D_];       // B[i][k] = bf16(Wgate[k][i])  (pre-transposed)
    __shared__ float  dphis[D_];

    if (tid < D_) dphis[tid] = dphibuf[(size_t)tb * D_ + tid];
    __syncthreads();
    #pragma unroll 4
    for (int idx = tid; idx < D_ * D_; idx += 256) {
        int k = idx & 127;
        As[idx] = f2bf(Wh[idx] * dphis[k]);
        Bs[idx] = WgT[idx];
    }
    __syncthreads();

    const int wave = tid >> 5;           // 0..7 -> row strip j in [wave*16, wave*16+16)
    const int lane = tid & 31;
    const int hi   = lane >> 4;          // 0/1
    const int nloc = lane & 15;

    const int ra = wave * 16 + nloc;     // A row for this lane (M = lane&15 within tile)
    // precompute per-lane i-independent scalars after GEMM

    #pragma unroll 1
    for (int nt = 0; nt < 8; ++nt) {
        v8f c = {0.f, 0.f, 0.f, 0.f, 0.f, 0.f, 0.f, 0.f};
        const int rb = nt * 16 + nloc;   // B "row" in WgT (= output column i)
        #pragma unroll
        for (int kc = 0; kc < D_; kc += 32) {
            // A: elems 0..7 = K kc+8*hi .. +8 ; elems 8..15 = K kc+16+8*hi .. +8
            const v8bf a0 = *(const v8bf*)&As[ra * D_ + kc + 8 * hi];
            const v8bf a1 = *(const v8bf*)&As[ra * D_ + kc + 16 + 8 * hi];
            const v8bf b0 = *(const v8bf*)&Bs[rb * D_ + kc + 8 * hi];
            const v8bf b1 = *(const v8bf*)&Bs[rb * D_ + kc + 16 + 8 * hi];
            v16bf av = __builtin_shufflevector(a0, a1, 0,1,2,3,4,5,6,7,8,9,10,11,12,13,14,15);
            v16bf bv = __builtin_shufflevector(b0, b1, 0,1,2,3,4,5,6,7,8,9,10,11,12,13,14,15);
            c = __builtin_amdgcn_wmma_f32_16x16x32_bf16(false, av, false, bv,
                                                        (short)0, c, false, false);
        }
        // store: lane owns output row i = nt*16+nloc, cols j = wave*16 + 8*hi + r
        const int i = nt * 16 + nloc;
        float g       = gbuf[(size_t)tb * D_ + i];
        float h_entry = (t == 0) ? h0[b * D_ + i]
                                 : y[((size_t)b * T_ + (t - 1)) * D_ + i];
        float coef = (h_entry - 1.0f) * g * (1.0f - g);
        const int jbase = wave * 16 + 8 * hi;
        float o[8];
        #pragma unroll
        for (int r = 0; r < 8; ++r) {
            float v = coef * c[r];
            if (jbase + r == i) v += g;
            o[r] = v;
        }
        float* dst = jac_out + ((size_t)tb * D_ + i) * D_ + jbase;
        *(float4*)(dst)     = make_float4(o[0], o[1], o[2], o[3]);
        *(float4*)(dst + 4) = make_float4(o[4], o[5], o[6], o[7]);
    }
}

// ---------- 5) dh_dWh[b,i,p,q] = coefF[b,i] * hF[b,p] * dphiF[b,q] * Wgate[q,i] ----------
__global__ void dwh_kernel(const float* __restrict__ Wgate, const float* __restrict__ hfin,
                           const float* __restrict__ coefF, const float* __restrict__ dphiF,
                           float* __restrict__ dwh_out) {
    const int bi = blockIdx.x;           // b*D + i   (4096 blocks)
    const int b  = bi >> 7;
    const int i  = bi & 127;
    const int tid = threadIdx.x;         // 0..255
    __shared__ float R[D_];
    __shared__ float hFs[D_];
    if (tid < D_) {
        R[tid]   = coefF[bi] * dphiF[b * D_ + tid] * Wgate[tid * D_ + i];
        hFs[tid] = hfin[b * D_ + tid];
    }
    __syncthreads();
    float* dst = dwh_out + (size_t)bi * (D_ * D_);
    #pragma unroll 4
    for (int idx = tid * 4; idx < D_ * D_; idx += 256 * 4) {
        int p = idx >> 7, q = idx & 127;
        float hp = hFs[p];
        *(float4*)(dst + idx) = make_float4(hp * R[q], hp * R[q + 1],
                                            hp * R[q + 2], hp * R[q + 3]);
    }
}

// ---------- launcher ----------
extern "C" void kernel_launch(void* const* d_in, const int* in_sizes, int n_in,
                              void* d_out, int out_size, void* d_ws, size_t ws_size,
                              hipStream_t stream) {
    (void)in_sizes; (void)n_in; (void)out_size; (void)ws_size;
    const float* x     = (const float*)d_in[0];   // (B,T,IN)
    const float* h0    = (const float*)d_in[1];   // (B,D)
    const float* Wx    = (const float*)d_in[2];   // (IN,D)
    const float* Wgate = (const float*)d_in[3];   // (D,D)
    const float* Wh    = (const float*)d_in[4];   // (D,D)

    float* out = (float*)d_out;
    const size_t OFF_H   = 0;
    const size_t OFF_Y   = (size_t)B_ * D_;                              // 4096
    const size_t OFF_JAC = OFF_Y + (size_t)B_ * T_ * D_;                 // 1052672
    const size_t OFF_DWH = OFF_JAC + (size_t)T_ * B_ * D_ * D_;          // 135270400
    float* hfin    = out + OFF_H;
    float* y       = out + OFF_Y;
    float* jac_out = out + OFF_JAC;
    float* dwh_out = out + OFF_DWH;

    char* ws = (char*)d_ws;
    const size_t SZ_TBD = (size_t)T_ * B_ * D_ * sizeof(float);          // 4 MiB
    float*  dphibuf = (float*)(ws);
    float*  gbuf    = (float*)(ws + SZ_TBD);
    __bf16* WgT     = (__bf16*)(ws + 2 * SZ_TBD);
    float*  coefF   = (float*)(ws + 2 * SZ_TBD + (size_t)D_ * D_ * 2);
    float*  dphiF   = coefF + (size_t)B_ * D_;

    prep_wgt_kernel<<<64, 256, 0, stream>>>(Wgate, WgT);
    fwd_kernel<<<B_, D_, 0, stream>>>(x, h0, Wx, Wgate, Wh, hfin, y, gbuf, dphibuf);
    finstep_kernel<<<B_, D_, 0, stream>>>(x, hfin, Wx, Wgate, Wh, coefF, dphiF);
    jac_kernel<<<T_ * B_, 256, 0, stream>>>(Wh, WgT, gbuf, dphibuf, y, h0, jac_out);
    dwh_kernel<<<B_ * D_, 256, 0, stream>>>(Wgate, hfin, coefF, dphiF, dwh_out);
}
